// BenchmarkSelfAttention_19241453486813
// MI455X (gfx1250) — matrix operand order repaired and checked by reading, not compiled, not run on hardware
//
#include <hip/hip_runtime.h>

#define DEV __device__ __forceinline__

typedef __attribute__((ext_vector_type(16))) _Float16 v16h;
typedef __attribute__((ext_vector_type(8)))  _Float16 v8h;
typedef __attribute__((ext_vector_type(4)))  _Float16 v4h;
typedef __attribute__((ext_vector_type(8)))  float    v8f;
typedef __attribute__((ext_vector_type(4)))  float    v4f;

union F16Frag { v16h v; v8h h2[2]; _Float16 e[16]; };

DEV v8f wmma16(v16h a, v16h b, v8f c) {
  // D = A(16x32 f16) * B(32x16 f16) + C(16x16 f32)
  return __builtin_amdgcn_wmma_f32_16x16x32_f16(false, a, false, b, (short)0, c,
                                                false, false);
}

// A-fragment (16x32 f16) load: lane L holds row m = L%16.
// half h -> k = ks*32 + h + 16*(L>=16)  (two contiguous 16B chunks)
DEV v16h load_a_frag(const _Float16* row, int ks, int hi) {
  F16Frag f;
  const _Float16* p = row + ks * 32 + hi * 16;
  f.h2[0] = *(const v8h*)(p);
  f.h2[1] = *(const v8h*)(p + 8);
  return f.v;
}

// ---------------------------------------------------------------------------
// GEMM: C[m,n] = sum_k X[m,k] * W[n,k] + bias[n]   (y = x @ W.T + b)
// X: 4096 x 1024 (f32 or f16), W: 1024 x 1024 f32 row-major.
// Block tile 128x64, wave tile 32x64 (2 A-frags share each B-frag).
// out_mode 0: f16 store permuted to [B,H,S,D]; out_mode 1: f32 store [m,n].
// ---------------------------------------------------------------------------
struct GemmLds {
  alignas(128) _Float16 A[128 * 64];  // [m][k]
  alignas(128) _Float16 B[64 * 64];   // [n][k] (B column-major == W row-major)
};

__global__ __launch_bounds__(128) void proj_gemm_kernel(
    const void* __restrict__ Xv, int x_is_f16,
    const float* __restrict__ W, const float* __restrict__ bias,
    void* __restrict__ Out, int out_mode)
{
  __shared__ GemmLds lds;
  const int m0 = blockIdx.x * 128;
  const int n0 = blockIdx.y * 64;
  const int tid = threadIdx.x;
  const int wave = tid >> 5, lane = tid & 31;
  const int hi = lane >> 4, ln = lane & 15;

  v8f acc[2][4] = {};

  for (int kt = 0; kt < 1024; kt += 64) {
    if (x_is_f16) {
      const _Float16* X16 = (const _Float16*)Xv;
      #pragma unroll
      for (int i = 0; i < 8; ++i) {
        int idx8 = tid + i * 128;            // 1024 chunks of 8 halves
        int row = idx8 >> 3, c8 = idx8 & 7;
        *(v8h*)(&lds.A[row * 64 + c8 * 8]) =
            *(const v8h*)(X16 + (size_t)(m0 + row) * 1024 + kt + c8 * 8);
      }
      if (kt + 64 < 1024)
        __builtin_prefetch(X16 + (size_t)(m0 + tid) * 1024 + kt + 64, 0, 1);
    } else {
      const float* X = (const float*)Xv;
      #pragma unroll
      for (int i = 0; i < 16; ++i) {
        int idx4 = tid + i * 128;            // 2048 chunks of 4 floats
        int row = idx4 >> 4, c4 = idx4 & 15;
        v4f fv = *(const v4f*)(X + (size_t)(m0 + row) * 1024 + kt + c4 * 4);
        v4h hv;
        hv[0] = (_Float16)fv[0]; hv[1] = (_Float16)fv[1];
        hv[2] = (_Float16)fv[2]; hv[3] = (_Float16)fv[3];
        *(v4h*)(&lds.A[row * 64 + c4 * 4]) = hv;
      }
      if (kt + 64 < 1024)
        __builtin_prefetch(X + (size_t)(m0 + tid) * 1024 + kt + 64, 0, 1);
    }
    #pragma unroll
    for (int i = 0; i < 8; ++i) {
      int idx4 = tid + i * 128;
      int row = idx4 >> 4, c4 = idx4 & 15;
      v4f fv = *(const v4f*)(W + (size_t)(n0 + row) * 1024 + kt + c4 * 4);
      v4h hv;
      hv[0] = (_Float16)fv[0]; hv[1] = (_Float16)fv[1];
      hv[2] = (_Float16)fv[2]; hv[3] = (_Float16)fv[3];
      *(v4h*)(&lds.B[row * 64 + c4 * 4]) = hv;
    }
    if (kt + 64 < 1024 && tid < 64)
      __builtin_prefetch(W + (size_t)(n0 + tid) * 1024 + kt + 64, 0, 1);
    __syncthreads();

    #pragma unroll
    for (int ks = 0; ks < 2; ++ks) {
      v16h a0 = load_a_frag(&lds.A[(wave * 32 + ln) * 64], ks, hi);
      v16h a1 = load_a_frag(&lds.A[(wave * 32 + 16 + ln) * 64], ks, hi);
      #pragma unroll
      for (int nt = 0; nt < 4; ++nt) {
        // B-frag: lane col n = ln, halves = contiguous k (16 per half-wave)
        v16h b = *(const v16h*)(&lds.B[(nt * 16 + ln) * 64 + ks * 32 + hi * 16]);
        acc[0][nt] = wmma16(a0, b, acc[0][nt]);
        acc[1][nt] = wmma16(a1, b, acc[1][nt]);
      }
    }
    __syncthreads();
  }

  #pragma unroll
  for (int rb = 0; rb < 2; ++rb) {
    #pragma unroll
    for (int nt = 0; nt < 4; ++nt) {
      #pragma unroll
      for (int r = 0; r < 8; ++r) {
        int m = m0 + wave * 32 + rb * 16 + r + 8 * hi;  // C layout: M = r (+8 hi)
        int n = n0 + nt * 16 + ln;
        float val = acc[rb][nt][r] + bias[n];
        if (out_mode == 0) {                 // f16, permute (b,s,h,d)->[B,H,S,D]
          int b = m >> 11, s = m & 2047;
          int h = n >> 6, d = n & 63;
          ((_Float16*)Out)[(((size_t)b * 16 + h) * 2048 + s) * 64 + d] =
              (_Float16)val;
        } else {                             // f32 plain [m,n]
          ((float*)Out)[(size_t)m * 1024 + n] = val;
        }
      }
    }
  }
}

// ---------------------------------------------------------------------------
// Flash attention: per (b,h) head, 128 queries/block (32/wave), key tiles of 64.
// ---------------------------------------------------------------------------
struct AttnLds {
  alignas(128) _Float16 K[64 * 64];     // [j][d]
  alignas(128) _Float16 Vt[64 * 64];    // transposed: [d][j]
  alignas(128) float S[4][32 * 64];     // per-wave score/prob tile
  float rmax[4][32];
  float rsum[4][32];
  float rscale[4][32];
};

__global__ __launch_bounds__(128) void attn_kernel(
    const _Float16* __restrict__ Qp, const _Float16* __restrict__ Kp,
    const _Float16* __restrict__ Vp, _Float16* __restrict__ Obuf)
{
  __shared__ AttnLds lds;
  const int bh = blockIdx.y;              // b*16 + h
  const int q0 = blockIdx.x * 128;
  const int tid = threadIdx.x, wave = tid >> 5, lane = tid & 31;
  const int hi = lane >> 4, ln = lane & 15;

  const size_t head_off = (size_t)bh * 2048 * 64;
  const _Float16* Qh = Qp + head_off;
  const _Float16* Kh = Kp + head_off;
  const _Float16* Vh = Vp + head_off;

  // Q fragments live in registers for the whole kernel (32 rows per wave, D=64)
  v16h qf[2][2];
  #pragma unroll
  for (int rb = 0; rb < 2; ++rb)
    #pragma unroll
    for (int ks = 0; ks < 2; ++ks)
      qf[rb][ks] =
          load_a_frag(Qh + (size_t)(q0 + wave * 32 + rb * 16 + ln) * 64, ks, hi);

  v8f oacc[2][4] = {};
  if (lane < 16) {
    #pragma unroll
    for (int rg = 0; rg < 2; ++rg) {
      lds.rmax[wave][rg * 16 + lane] = -1e30f;
      lds.rsum[wave][rg * 16 + lane] = 0.0f;
    }
  }

  for (int j0 = 0; j0 < 2048; j0 += 64) {
    // Stage K tile row-major and V tile transposed
    #pragma unroll
    for (int i = 0; i < 4; ++i) {
      int idx8 = tid + i * 128;
      int row = idx8 >> 3, c8 = idx8 & 7;
      *(v8h*)(&lds.K[row * 64 + c8 * 8]) =
          *(const v8h*)(Kh + (size_t)(j0 + row) * 64 + c8 * 8);
      v8h vv = *(const v8h*)(Vh + (size_t)(j0 + row) * 64 + c8 * 8);
      #pragma unroll
      for (int e = 0; e < 8; ++e) lds.Vt[(c8 * 8 + e) * 64 + row] = vv[e];
    }
    if (j0 + 64 < 2048) {
      __builtin_prefetch(Kh + (size_t)(j0 + 64) * 64 + tid * 32, 0, 1);
      __builtin_prefetch(Vh + (size_t)(j0 + 64) * 64 + tid * 32, 0, 1);
    }
    __syncthreads();

    // scores = Q @ K^T * D^-0.5 : each K B-frag feeds two row-block WMMAs
    v8f sacc[2][4] = {};
    #pragma unroll
    for (int ks = 0; ks < 2; ++ks) {
      #pragma unroll
      for (int nt = 0; nt < 4; ++nt) {
        v16h b = *(const v16h*)(&lds.K[(nt * 16 + ln) * 64 + ks * 32 + hi * 16]);
        sacc[0][nt] = wmma16(qf[0][ks], b, sacc[0][nt]);
        sacc[1][nt] = wmma16(qf[1][ks], b, sacc[1][nt]);
      }
    }
    #pragma unroll
    for (int rb = 0; rb < 2; ++rb)
      #pragma unroll
      for (int nt = 0; nt < 4; ++nt)
        #pragma unroll
        for (int r = 0; r < 8; ++r)
          lds.S[wave][(rb * 16 + r + 8 * hi) * 64 + nt * 16 + ln] =
              sacc[rb][nt][r] * 0.125f;
    __syncthreads();

    // online softmax: each lane owns half of 2 rows (32 cols), combine via shfl
    #pragma unroll
    for (int rg = 0; rg < 2; ++rg) {
      const int row = rg * 16 + ln;
      float* srow = &lds.S[wave][row * 64];
      const int cb = hi * 32;
      float tmax = -1e30f;
      #pragma unroll
      for (int c = 0; c < 32; ++c) tmax = fmaxf(tmax, srow[cb + c]);
      tmax = fmaxf(tmax, __shfl_xor(tmax, 16, 32));
      float oldm = lds.rmax[wave][row];
      float nm = fmaxf(oldm, tmax);
      float psum = 0.0f;
      #pragma unroll
      for (int c = 0; c < 32; ++c) {
        float p = __expf(srow[cb + c] - nm);
        srow[cb + c] = p;
        psum += p;
      }
      psum += __shfl_xor(psum, 16, 32);
      float fsc = __expf(oldm - nm);
      if (hi == 0) {
        lds.rmax[wave][row] = nm;
        lds.rsum[wave][row] = lds.rsum[wave][row] * fsc + psum;
        lds.rscale[wave][row] = fsc;
      }
    }
    __syncthreads();

    // rescale running output, then O += P @ V
    #pragma unroll
    for (int rb = 0; rb < 2; ++rb)
      #pragma unroll
      for (int nt = 0; nt < 4; ++nt)
        #pragma unroll
        for (int r = 0; r < 8; ++r)
          oacc[rb][nt][r] *= lds.rscale[wave][rb * 16 + r + 8 * hi];

    #pragma unroll
    for (int ks = 0; ks < 2; ++ks) {
      F16Frag pf0, pf1;
      #pragma unroll
      for (int h = 0; h < 16; ++h) {
        int k = ks * 32 + h + 16 * hi;  // A-layout pattern
        pf0.e[h] = (_Float16)lds.S[wave][ln * 64 + k];
        pf1.e[h] = (_Float16)lds.S[wave][(16 + ln) * 64 + k];
      }
      #pragma unroll
      for (int nt = 0; nt < 4; ++nt) {
        v16h b = *(const v16h*)(&lds.Vt[(nt * 16 + ln) * 64 + ks * 32 + hi * 16]);
        oacc[0][nt] = wmma16(pf0.v, b, oacc[0][nt]);
        oacc[1][nt] = wmma16(pf1.v, b, oacc[1][nt]);
      }
    }
    __syncthreads();
  }

  // normalize and store O as f16 in [B, S, H*D] (ready to be GEMM-A for wO)
  const int bb = bh >> 4, hh = bh & 15;
  #pragma unroll
  for (int rb = 0; rb < 2; ++rb) {
    #pragma unroll
    for (int nt = 0; nt < 4; ++nt) {
      #pragma unroll
      for (int r = 0; r < 8; ++r) {
        int mrow = rb * 16 + r + 8 * hi;
        int s = q0 + wave * 32 + mrow;
        float inv = 1.0f / lds.rsum[wave][mrow];
        float val = oacc[rb][nt][r] * inv;
        Obuf[((size_t)bb * 2048 + s) * 1024 + hh * 64 + nt * 16 + ln] =
            (_Float16)val;
      }
    }
  }
}

// ---------------------------------------------------------------------------
extern "C" void kernel_launch(void* const* d_in, const int* in_sizes, int n_in,
                              void* d_out, int out_size, void* d_ws, size_t ws_size,
                              hipStream_t stream)
{
  (void)in_sizes; (void)n_in; (void)out_size; (void)ws_size;
  const float* Q  = (const float*)d_in[0];
  const float* K  = (const float*)d_in[1];
  const float* V  = (const float*)d_in[2];
  const float* wQ = (const float*)d_in[3];
  const float* bQ = (const float*)d_in[4];
  const float* wK = (const float*)d_in[5];
  const float* bK = (const float*)d_in[6];
  const float* wV = (const float*)d_in[7];
  const float* bV = (const float*)d_in[8];
  const float* wO = (const float*)d_in[9];
  const float* bO = (const float*)d_in[10];

  const size_t HELEMS = (size_t)2 * 16 * 2048 * 64;  // 4,194,304 f16 per tensor
  _Float16* Qp = (_Float16*)d_ws;
  _Float16* Kp = Qp + HELEMS;
  _Float16* Vp = Kp + HELEMS;
  _Float16* Ob = Vp + HELEMS;                         // [B,S,M] f16

  dim3 blk(128);
  dim3 gridG(4096 / 128, 1024 / 64);                  // 32 x 16 tiles
  proj_gemm_kernel<<<gridG, blk, 0, stream>>>(Q, 0, wQ, bQ, Qp, 0);
  proj_gemm_kernel<<<gridG, blk, 0, stream>>>(K, 0, wK, bK, Kp, 0);
  proj_gemm_kernel<<<gridG, blk, 0, stream>>>(V, 0, wV, bV, Vp, 0);

  dim3 gridA(2048 / 128, 2 * 16);                     // 16 query blocks x 32 heads
  attn_kernel<<<gridA, blk, 0, stream>>>(Qp, Kp, Vp, Ob);

  proj_gemm_kernel<<<gridG, blk, 0, stream>>>(Ob, 1, wO, bO, d_out, 1);
}
